// QRoPETRegressor_90726889161283
// MI455X (gfx1250) — matrix operand order repaired and checked by reading, not compiled, not run on hardware
//
#include <hip/hip_runtime.h>
#include <hip/hip_bf16.h>
#include <stdint.h>

typedef __bf16 bf16_t;
typedef __attribute__((ext_vector_type(16))) __bf16 v16bf;
typedef __attribute__((ext_vector_type(8)))  __bf16 v8bf;
typedef __attribute__((ext_vector_type(8)))  float  v8f;

#define Bn   32
#define SEQ  520
#define DIM  512
#define NHD  8
#define HD   64
#define MLPD 2048
#define NL   4
#define NO   8
#define NF   16
#define MROWS (Bn * SEQ)      // 16640 = 130 * 128
#define SPAD  640             // padded seq (mult of 128 and 32)

__device__ __forceinline__ bf16_t f2bf(float x) { return (bf16_t)x; }
__device__ __forceinline__ float softplusf(float x) { return log1pf(expf(x)); }

__device__ __forceinline__ v16bf join8(v8bf lo, v8bf hi) {
    return __builtin_shufflevector(lo, hi, 0,1,2,3,4,5,6,7,8,9,10,11,12,13,14,15);
}

// async global->LDS copy, 16B per lane, tracked by ASYNCcnt
__device__ __forceinline__ void async_ld_b128(uint32_t lds_off, const bf16_t* gptr) {
    asm volatile("global_load_async_to_lds_b128 %0, %1, off"
                 :: "v"(lds_off), "v"(gptr) : "memory");
}
__device__ __forceinline__ void wait_async0() {
    asm volatile("s_wait_asynccnt 0x0" ::: "memory");
}
__device__ __forceinline__ uint32_t lds_off32(const void* p) {
    return (uint32_t)(uintptr_t)p;
}

// ---------------------------------------------------------------------------
// elementwise helpers
// ---------------------------------------------------------------------------
__global__ void f2bf_kernel(const float* __restrict__ src, bf16_t* __restrict__ dst, size_t n) {
    size_t i = (size_t)blockIdx.x * blockDim.x + threadIdx.x;
    if (i < n) dst[i] = f2bf(src[i]);
}

__global__ void zero_bf16_kernel(bf16_t* __restrict__ p, size_t n) {
    size_t i = (size_t)blockIdx.x * blockDim.x + threadIdx.x;
    if (i < n) p[i] = f2bf(0.f);
}

// ---------------------------------------------------------------------------
// input projection: h[b,s,:] = (s<8) ? cls[s,:] : x[b,s-8,:] @ W_in + b_in
// ---------------------------------------------------------------------------
__global__ void input_proj_kernel(const float* __restrict__ x,
                                  const float* __restrict__ W_in,
                                  const float* __restrict__ b_in,
                                  const float* __restrict__ cls,
                                  float* __restrict__ hf, bf16_t* __restrict__ hb) {
    size_t idx = (size_t)blockIdx.x * blockDim.x + threadIdx.x;
    if (idx >= (size_t)MROWS * DIM) return;
    int d = idx % DIM;
    int s = (idx / DIM) % SEQ;
    int b = idx / ((size_t)DIM * SEQ);
    float v;
    if (s < NO) {
        v = cls[s * DIM + d];
    } else {
        const float* xr = x + ((size_t)b * (SEQ - NO) + (s - NO)) * NF;
        v = b_in[d];
        #pragma unroll
        for (int f = 0; f < NF; ++f) v = fmaf(xr[f], W_in[f * DIM + d], v);
    }
    hf[idx] = v;
    hb[idx] = f2bf(v);
}

// ---------------------------------------------------------------------------
// generic bf16 WMMA GEMM: C[M,N] = A[M,K] @ W[K,N] + bias
// block 256 (8 waves), tile 128x64, K-step 32, double-buffered:
//   A tile : async global->LDS (row-major), ASYNCcnt pipelined
//   W tile : global->reg->LDS transposed (Wt[n][k]) so B-frags are contiguous
// ---------------------------------------------------------------------------
#define BM 128
#define BN 64
#define BK 32

template<bool RELU, bool OUTF, bool OUTB>
__global__ __launch_bounds__(256)
void gemm_bf16_kernel(const bf16_t* __restrict__ A, const bf16_t* __restrict__ W,
                      const float* __restrict__ bias,
                      float* __restrict__ Cf, bf16_t* __restrict__ Cb,
                      int M, int N, int K) {
    __shared__ __attribute__((aligned(16))) bf16_t As[2][BM][BK];  // 2 x 8 KB
    __shared__ __attribute__((aligned(16))) bf16_t Wt[2][BN][BK];  // 2 x 4 KB (transposed)

    const int tile_m = blockIdx.x;
    const int tile_n = blockIdx.y;
    const int tid  = threadIdx.x;
    const int wave = tid >> 5;
    const int lane = tid & 31;
    const int wm = wave >> 1;       // 0..3
    const int wn = wave & 1;        // 0..1

    v8f acc[2][2] = {};

    const int a_row  = lane & 15;
    const int a_kgrp = (lane >> 4) * 8;
    const int b_col  = lane & 15;
    const int b_kgrp = (lane >> 4) * 8;

    const int NK = K / BK;

    // staging helpers (per-thread work items)
    const int a_r0 = tid >> 2, a_c0 = (tid & 3) * 8;          // chunk tid
    const int a_r1 = (tid + 256) >> 2, a_c1 = ((tid + 256) & 3) * 8;
    const int w_r  = tid >> 3, w_c = (tid & 7) * 8;

    // ---- prologue: stage tile 0 into buffer 0 ----
    async_ld_b128(lds_off32(&As[0][a_r0][a_c0]),
                  A + (size_t)(tile_m * BM + a_r0) * K + a_c0);
    async_ld_b128(lds_off32(&As[0][a_r1][a_c1]),
                  A + (size_t)(tile_m * BM + a_r1) * K + a_c1);
    {
        v8bf w = *(const v8bf*)(W + (size_t)w_r * N + tile_n * BN + w_c);
        #pragma unroll
        for (int e = 0; e < 8; ++e) Wt[0][w_c + e][w_r] = w[e];
    }
    wait_async0();
    __syncthreads();

    for (int kt = 0; kt < NK; ++kt) {
        const int cur = kt & 1, nb = cur ^ 1;
        const bool pre = (kt + 1) < NK;
        v8bf wreg;
        if (pre) {
            const int k0n = (kt + 1) * BK;
            async_ld_b128(lds_off32(&As[nb][a_r0][a_c0]),
                          A + (size_t)(tile_m * BM + a_r0) * K + k0n + a_c0);
            async_ld_b128(lds_off32(&As[nb][a_r1][a_c1]),
                          A + (size_t)(tile_m * BM + a_r1) * K + k0n + a_c1);
            wreg = *(const v8bf*)(W + (size_t)(k0n + w_r) * N + tile_n * BN + w_c);
        }

        // ---- fragments: all contiguous 16B LDS loads ----
        v16bf afrag[2], bfrag[2];
        #pragma unroll
        for (int mi = 0; mi < 2; ++mi) {
            const bf16_t* base = &As[cur][wm * 32 + mi * 16 + a_row][0];
            afrag[mi] = join8(*(const v8bf*)(base + a_kgrp),
                              *(const v8bf*)(base + a_kgrp + 16));
        }
        #pragma unroll
        for (int ni = 0; ni < 2; ++ni) {
            const bf16_t* base = &Wt[cur][wn * 32 + ni * 16 + b_col][0];
            bfrag[ni] = join8(*(const v8bf*)(base + b_kgrp),
                              *(const v8bf*)(base + b_kgrp + 16));
        }
        #pragma unroll
        for (int mi = 0; mi < 2; ++mi)
            #pragma unroll
            for (int ni = 0; ni < 2; ++ni)
                acc[mi][ni] = __builtin_amdgcn_wmma_f32_16x16x32_bf16(
                    false, afrag[mi], false, bfrag[ni], (short)0, acc[mi][ni], false, false);

        if (pre) {
            #pragma unroll
            for (int e = 0; e < 8; ++e) Wt[nb][w_c + e][w_r] = wreg[e];
        }
        wait_async0();
        __syncthreads();
    }

    const int c_half = (lane >> 4) * 8;
    const int c_col  = lane & 15;
    #pragma unroll
    for (int mi = 0; mi < 2; ++mi) {
        #pragma unroll
        for (int ni = 0; ni < 2; ++ni) {
            int col = tile_n * BN + wn * 32 + ni * 16 + c_col;
            float bv = bias[col];
            #pragma unroll
            for (int r = 0; r < 8; ++r) {
                int row = tile_m * BM + wm * 32 + mi * 16 + c_half + r;
                float v = acc[mi][ni][r] + bv;
                if (RELU) v = fmaxf(v, 0.f);
                size_t o = (size_t)row * N + col;
                if (OUTF) Cf[o] = v;
                if (OUTB) Cb[o] = f2bf(v);
            }
        }
    }
}

// ---------------------------------------------------------------------------
// RoPE on q,k (full-D pairwise rotation, matching reference) + head split to
// [B*NH, SPAD, HD] bf16 buffers (pad rows pre-zeroed).
// NOTE: 1/sqrt(HD) = 0.125 is folded into q here (exact power-of-two scale),
// so the attention kernel needs no post-WMMA score scaling.
// ---------------------------------------------------------------------------
__global__ void rope_split_kernel(const float* __restrict__ qkv,
                                  bf16_t* __restrict__ qh, bf16_t* __restrict__ kh,
                                  bf16_t* __restrict__ vh) {
    size_t idx = (size_t)blockIdx.x * blockDim.x + threadIdx.x;
    const int halfD = DIM / 2;
    if (idx >= (size_t)Bn * SEQ * halfD) return;
    int i = idx % halfD;
    int s = (idx / halfD) % SEQ;
    int b = idx / ((size_t)halfD * SEQ);

    const float* row = qkv + ((size_t)b * SEQ + s) * (3 * DIM);
    float theta = (float)s * expf(-((float)(2 * i) / (float)DIM) * 9.210340371976184f);
    float sn = sinf(theta), cs = cosf(theta);

    int h  = (2 * i) / HD;
    int hd = (2 * i) - h * HD;
    size_t o = ((size_t)(b * NHD + h) * SPAD + s) * HD + hd;

    const float QS = 0.125f;   // 1/sqrt(HD), folded into q
    float q1 = row[2 * i], q2 = row[2 * i + 1];
    qh[o]     = f2bf((q1 * cs - q2 * sn) * QS);
    qh[o + 1] = f2bf((q1 * sn + q2 * cs) * QS);

    float k1 = row[DIM + 2 * i], k2 = row[DIM + 2 * i + 1];
    kh[o]     = f2bf(k1 * cs - k2 * sn);
    kh[o + 1] = f2bf(k1 * sn + k2 * cs);

    vh[o]     = f2bf(row[2 * DIM + 2 * i]);
    vh[o + 1] = f2bf(row[2 * DIM + 2 * i + 1]);
}

// ---------------------------------------------------------------------------
// flash attention: 256 thr = 8 waves; each wave owns 16 query rows.
// K staged async (row-major), V staged transposed; double-buffered.
// Key loop covers only ceil(SEQ/32)=17 tiles; masking only in the last tile.
// grid.x = B*NH * (SPAD/128)
// ---------------------------------------------------------------------------
#define NKV ((SEQ + 31) / 32)   // 17

__global__ __launch_bounds__(256)
void attention_kernel(const bf16_t* __restrict__ qh, const bf16_t* __restrict__ kh,
                      const bf16_t* __restrict__ vh, bf16_t* __restrict__ outb) {
    const int qBlocks = SPAD / 128;                 // 5
    const int bh   = blockIdx.x / qBlocks;
    const int qblk = blockIdx.x % qBlocks;
    const int b = bh / NHD, h = bh % NHD;

    const int tid  = threadIdx.x;
    const int wave = tid >> 5;
    const int lane = tid & 31;
    const int half = lane >> 4;
    const int l16  = lane & 15;

    __shared__ __attribute__((aligned(16))) bf16_t Ks[2][32][HD];  // 2 x 4 KB
    __shared__ __attribute__((aligned(16))) bf16_t Vt[2][HD][32];  // 2 x 4 KB (transposed)
    __shared__ __attribute__((aligned(16))) bf16_t Ps[8][16][32];  // 8 KB

    const int qrow0 = qblk * 128 + wave * 16;
    const size_t bhbase = (size_t)bh * SPAD;

    // load Q A-fragments (16 x 64 -> 2 chunks of K=32) from global, contiguous
    v16bf aq[2];
    {
        const int kgrp = half * 8;
        const bf16_t* qp = qh + (bhbase + qrow0 + l16) * HD;
        #pragma unroll
        for (int c = 0; c < 2; ++c)
            aq[c] = join8(*(const v8bf*)(qp + c * 32 + kgrp),
                          *(const v8bf*)(qp + c * 32 + kgrp + 16));
    }

    v8f acc[4] = {};
    float mrow[8], lrow[8];
    #pragma unroll
    for (int r = 0; r < 8; ++r) { mrow[r] = -1e30f; lrow[r] = 0.f; }

    const int b_kgrp = half * 8;    // B-frag K group
    const int a_kgrp = half * 8;    // A-frag K group (for P)

    const int kv_r = tid >> 3, kv_c = (tid & 7) * 8;   // 256 chunks of 8 elems

    // ---- prologue: stage tile 0 ----
    async_ld_b128(lds_off32(&Ks[0][kv_r][kv_c]), kh + (bhbase + kv_r) * HD + kv_c);
    {
        v8bf v = *(const v8bf*)(vh + (bhbase + kv_r) * HD + kv_c);
        #pragma unroll
        for (int e = 0; e < 8; ++e) Vt[0][kv_c + e][kv_r] = v[e];
    }
    wait_async0();
    __syncthreads();

    for (int kt = 0; kt < NKV; ++kt) {
        const int cur = kt & 1, nb = cur ^ 1;
        const bool pre = (kt + 1) < NKV;
        v8bf vreg;
        if (pre) {
            const size_t nbase = bhbase + (kt + 1) * 32;
            async_ld_b128(lds_off32(&Ks[nb][kv_r][kv_c]), kh + (nbase + kv_r) * HD + kv_c);
            vreg = *(const v8bf*)(vh + (nbase + kv_r) * HD + kv_c);
        }

        // ---- scores for two 16-key subtiles (q pre-scaled by 1/sqrt(HD)) ----
        float sc[2][8];
        #pragma unroll
        for (int jt = 0; jt < 2; ++jt) {
            const bf16_t* kp = &Ks[cur][jt * 16 + l16][0];
            v16bf bk0 = join8(*(const v8bf*)(kp + b_kgrp),      *(const v8bf*)(kp + b_kgrp + 16));
            v16bf bk1 = join8(*(const v8bf*)(kp + 32 + b_kgrp), *(const v8bf*)(kp + 32 + b_kgrp + 16));
            v8f s = {};
            s = __builtin_amdgcn_wmma_f32_16x16x32_bf16(false, aq[0], false, bk0, (short)0, s, false, false);
            s = __builtin_amdgcn_wmma_f32_16x16x32_bf16(false, aq[1], false, bk1, (short)0, s, false, false);
            if (kt == NKV - 1) {            // uniform branch: only last tile straddles SEQ
                bool valid = (kt * 32 + jt * 16 + l16) < SEQ;
                #pragma unroll
                for (int r = 0; r < 8; ++r) sc[jt][r] = valid ? s[r] : -1e30f;
            } else {
                #pragma unroll
                for (int r = 0; r < 8; ++r) sc[jt][r] = s[r];
            }
        }

        // ---- online softmax stats (row reductions across 16-lane halves) ----
        float mt[8];
        #pragma unroll
        for (int r = 0; r < 8; ++r) mt[r] = fmaxf(sc[0][r], sc[1][r]);
        #pragma unroll
        for (int off = 8; off >= 1; off >>= 1)
            #pragma unroll
            for (int r = 0; r < 8; ++r) mt[r] = fmaxf(mt[r], __shfl_xor(mt[r], off, 32));

        float alpha[8], psum[8];
        #pragma unroll
        for (int r = 0; r < 8; ++r) {
            float mnew = fmaxf(mrow[r], mt[r]);
            alpha[r] = __expf(mrow[r] - mnew);
            sc[0][r] = __expf(sc[0][r] - mnew);
            sc[1][r] = __expf(sc[1][r] - mnew);
            psum[r]  = sc[0][r] + sc[1][r];
            mrow[r]  = mnew;
        }
        #pragma unroll
        for (int off = 8; off >= 1; off >>= 1)
            #pragma unroll
            for (int r = 0; r < 8; ++r) psum[r] += __shfl_xor(psum[r], off, 32);
        #pragma unroll
        for (int r = 0; r < 8; ++r) lrow[r] = lrow[r] * alpha[r] + psum[r];

        // ---- write P to per-wave LDS and reload in A-fragment layout ----
        #pragma unroll
        for (int r = 0; r < 8; ++r) {
            int rl = r + half * 8;
            Ps[wave][rl][l16]      = f2bf(sc[0][r]);
            Ps[wave][rl][16 + l16] = f2bf(sc[1][r]);
        }
        asm volatile("s_wait_dscnt 0x0" ::: "memory");

        const bf16_t* pp = &Ps[wave][l16][0];
        v16bf ap = join8(*(const v8bf*)(pp + a_kgrp),
                         *(const v8bf*)(pp + a_kgrp + 16));

        // ---- rescale accumulators, then acc += P @ V ----
        #pragma unroll
        for (int ni = 0; ni < 4; ++ni)
            #pragma unroll
            for (int r = 0; r < 8; ++r) acc[ni][r] *= alpha[r];

        #pragma unroll
        for (int ni = 0; ni < 4; ++ni) {
            const bf16_t* vp = &Vt[cur][ni * 16 + l16][0];
            v16bf bv = join8(*(const v8bf*)(vp + b_kgrp),
                             *(const v8bf*)(vp + b_kgrp + 16));
            acc[ni] = __builtin_amdgcn_wmma_f32_16x16x32_bf16(
                false, ap, false, bv, (short)0, acc[ni], false, false);
        }

        if (pre) {
            #pragma unroll
            for (int e = 0; e < 8; ++e) Vt[nb][kv_c + e][kv_r] = vreg[e];
        }
        wait_async0();
        __syncthreads();
    }

    // ---- normalize and store merged-head output (valid rows only) ----
    float inv[8];
    #pragma unroll
    for (int r = 0; r < 8; ++r) inv[r] = 1.f / lrow[r];
    #pragma unroll
    for (int ni = 0; ni < 4; ++ni)
        #pragma unroll
        for (int r = 0; r < 8; ++r) {
            int srow = qrow0 + r + half * 8;
            if (srow < SEQ)
                outb[((size_t)b * SEQ + srow) * DIM + h * HD + ni * 16 + l16] =
                    f2bf(acc[ni][r] * inv[r]);
        }
}

// ---------------------------------------------------------------------------
// residual add + layernorm (one 512-elem row per block of 256 threads)
// ---------------------------------------------------------------------------
__global__ __launch_bounds__(256)
void add_ln_kernel(const float* __restrict__ x, const float* __restrict__ y,
                   const float* __restrict__ g, const float* __restrict__ be,
                   float* __restrict__ outf, bf16_t* __restrict__ outb) {
    size_t base = (size_t)blockIdx.x * DIM;
    int t = threadIdx.x;
    __shared__ float red[8];

    float v0 = x[base + t] + y[base + t];
    float v1 = x[base + t + 256] + y[base + t + 256];

    float s = v0 + v1;
    #pragma unroll
    for (int off = 16; off >= 1; off >>= 1) s += __shfl_xor(s, off, 32);
    if ((t & 31) == 0) red[t >> 5] = s;
    __syncthreads();
    float mean = 0.f;
    #pragma unroll
    for (int i = 0; i < 8; ++i) mean += red[i];
    mean *= (1.f / DIM);
    __syncthreads();

    float d0 = v0 - mean, d1 = v1 - mean;
    float vs = d0 * d0 + d1 * d1;
    #pragma unroll
    for (int off = 16; off >= 1; off >>= 1) vs += __shfl_xor(vs, off, 32);
    if ((t & 31) == 0) red[t >> 5] = vs;
    __syncthreads();
    float var = 0.f;
    #pragma unroll
    for (int i = 0; i < 8; ++i) var += red[i];
    var *= (1.f / DIM);

    float rs = rsqrtf(var + 1e-6f);
    float o0 = d0 * rs * g[t]       + be[t];
    float o1 = d1 * rs * g[t + 256] + be[t + 256];
    outf[base + t]       = o0;
    outf[base + t + 256] = o1;
    outb[base + t]       = f2bf(o0);
    outb[base + t + 256] = f2bf(o1);
}

// ---------------------------------------------------------------------------
// per-cls-token head: hid[b,o,j] = relu(sum_d h[b,o,d] * Wh[o,d,j] + bh[o,j])
// ---------------------------------------------------------------------------
__global__ void head_kernel(const float* __restrict__ hf, const float* __restrict__ Wh,
                            const float* __restrict__ bhp, float* __restrict__ hid) {
    size_t idx = (size_t)blockIdx.x * blockDim.x + threadIdx.x;
    if (idx >= (size_t)Bn * NO * DIM) return;
    int j = idx % DIM;
    int o = (idx / DIM) % NO;
    int b = idx / ((size_t)DIM * NO);
    const float* xr = hf + ((size_t)b * SEQ + o) * DIM;
    const float* w  = Wh + (size_t)o * DIM * DIM + j;
    float a = bhp[o * DIM + j];
    for (int d = 0; d < DIM; ++d) a = fmaf(xr[d], w[(size_t)d * DIM], a);
    hid[idx] = fmaxf(a, 0.f);
}

// ---------------------------------------------------------------------------
// variational heads: mu, lv  (256 outputs each)
// ---------------------------------------------------------------------------
__global__ void varhead_kernel(const float* __restrict__ hid,
                               const float* __restrict__ mu_wmu, const float* __restrict__ mu_wrho,
                               const float* __restrict__ mu_bmu, const float* __restrict__ mu_brho,
                               const float* __restrict__ lv_wmu, const float* __restrict__ lv_wrho,
                               const float* __restrict__ lv_bmu, const float* __restrict__ lv_brho,
                               const float* __restrict__ eps_mu_w, const float* __restrict__ eps_mu_b,
                               const float* __restrict__ eps_lv_w, const float* __restrict__ eps_lv_b,
                               float* __restrict__ out) {
    int idx = blockIdx.x * blockDim.x + threadIdx.x;
    if (idx >= Bn * NO) return;
    int o = idx % NO;
    const float* hr = hid + (size_t)idx * DIM;
    const float EPSc = 1e-8f;
    float amu = 0.f, alv = 0.f;
    for (int k = 0; k < DIM; ++k) {
        float hv = hr[k];
        int wi = o * DIM + k;
        float wmu = mu_wmu[wi] + (softplusf(mu_wrho[wi]) + EPSc) * eps_mu_w[wi];
        float wlv = lv_wmu[wi] + (softplusf(lv_wrho[wi]) + EPSc) * eps_lv_w[wi];
        amu = fmaf(hv, wmu, amu);
        alv = fmaf(hv, wlv, alv);
    }
    amu += mu_bmu[o] + (softplusf(mu_brho[o]) + EPSc) * eps_mu_b[o];
    alv += lv_bmu[o] + (softplusf(lv_brho[o]) + EPSc) * eps_lv_b[o];
    out[idx]           = amu;
    out[Bn * NO + idx] = logf(EPSc + softplusf(alv));
}

// ---------------------------------------------------------------------------
// host orchestration
// ---------------------------------------------------------------------------
static inline dim3 grid1(size_t n, int bs) { return dim3((unsigned)((n + bs - 1) / bs)); }

extern "C" void kernel_launch(void* const* d_in, const int* in_sizes, int n_in,
                              void* d_out, int out_size, void* d_ws, size_t ws_size,
                              hipStream_t stream) {
    const float* x      = (const float*)d_in[0];
    const float* W_in   = (const float*)d_in[1];
    const float* b_in   = (const float*)d_in[2];
    const float* cls    = (const float*)d_in[3];
    const float* Wqkv   = (const float*)d_in[4];
    const float* bqkv   = (const float*)d_in[5];
    const float* Wo     = (const float*)d_in[6];
    const float* bo     = (const float*)d_in[7];
    const float* ln1g   = (const float*)d_in[8];
    const float* ln1b   = (const float*)d_in[9];
    const float* Wm1    = (const float*)d_in[10];
    const float* bm1    = (const float*)d_in[11];
    const float* Wm2    = (const float*)d_in[12];
    const float* bm2    = (const float*)d_in[13];
    const float* ln2g   = (const float*)d_in[14];
    const float* ln2b   = (const float*)d_in[15];
    const float* Wh     = (const float*)d_in[16];
    const float* bhp    = (const float*)d_in[17];
    const float* mu_wmu = (const float*)d_in[18];
    const float* mu_wrho= (const float*)d_in[19];
    const float* mu_bmu = (const float*)d_in[20];
    const float* mu_brho= (const float*)d_in[21];
    const float* lv_wmu = (const float*)d_in[22];
    const float* lv_wrho= (const float*)d_in[23];
    const float* lv_bmu = (const float*)d_in[24];
    const float* lv_brho= (const float*)d_in[25];
    const float* eps_mu_w = (const float*)d_in[26];
    const float* eps_mu_b = (const float*)d_in[27];
    const float* eps_lv_w = (const float*)d_in[28];
    const float* eps_lv_b = (const float*)d_in[29];
    float* out = (float*)d_out;

    // ---- carve workspace ----
    char* wsb = (char*)d_ws;
    size_t off = 0;
    auto carve = [&](size_t bytes) -> void* {
        off = (off + 255) & ~(size_t)255;
        void* p = wsb + off;
        off += bytes;
        return p;
    };
    float*  hf    = (float*) carve((size_t)MROWS * DIM * 4);
    bf16_t* hb    = (bf16_t*)carve((size_t)MROWS * DIM * 2);
    float*  tmpf  = (float*) carve((size_t)MROWS * DIM * 4);
    // union: qkv fp32 (M x 1536 x 4B) overlaps mlp-hidden bf16 (M x 2048 x 2B)
    void*   big   = carve((size_t)MROWS * (3 * DIM) * 4);
    float*  qkvf  = (float*)big;
    bf16_t* m1b   = (bf16_t*)big;
    bf16_t* aob   = (bf16_t*)carve((size_t)MROWS * DIM * 2);
    bf16_t* qhb   = (bf16_t*)carve((size_t)Bn * NHD * SPAD * HD * 2);
    bf16_t* khb   = (bf16_t*)carve((size_t)Bn * NHD * SPAD * HD * 2);
    bf16_t* vhb   = (bf16_t*)carve((size_t)Bn * NHD * SPAD * HD * 2);
    bf16_t* Wqkvb = (bf16_t*)carve((size_t)NL * DIM * 3 * DIM * 2);
    bf16_t* Wob   = (bf16_t*)carve((size_t)NL * DIM * DIM * 2);
    bf16_t* Wm1b  = (bf16_t*)carve((size_t)NL * DIM * MLPD * 2);
    bf16_t* Wm2b  = (bf16_t*)carve((size_t)NL * MLPD * DIM * 2);
    float*  hid   = (float*) carve((size_t)Bn * NO * DIM * 4);

    // ---- weights -> bf16 ----
    {
        size_t n;
        n = (size_t)NL * DIM * 3 * DIM; f2bf_kernel<<<grid1(n, 256), 256, 0, stream>>>(Wqkv, Wqkvb, n);
        n = (size_t)NL * DIM * DIM;     f2bf_kernel<<<grid1(n, 256), 256, 0, stream>>>(Wo,   Wob,   n);
        n = (size_t)NL * DIM * MLPD;    f2bf_kernel<<<grid1(n, 256), 256, 0, stream>>>(Wm1,  Wm1b,  n);
        n = (size_t)NL * MLPD * DIM;    f2bf_kernel<<<grid1(n, 256), 256, 0, stream>>>(Wm2,  Wm2b,  n);
    }
    // zero q/k/v (pad rows must read as 0 every call)
    {
        size_t n = (size_t)Bn * NHD * SPAD * HD;
        zero_bf16_kernel<<<grid1(n, 256), 256, 0, stream>>>(qhb, n);
        zero_bf16_kernel<<<grid1(n, 256), 256, 0, stream>>>(khb, n);
        zero_bf16_kernel<<<grid1(n, 256), 256, 0, stream>>>(vhb, n);
    }

    // ---- embed + cls concat ----
    input_proj_kernel<<<grid1((size_t)MROWS * DIM, 256), 256, 0, stream>>>(
        x, W_in, b_in, cls, hf, hb);

    // ---- transformer layers ----
    for (int l = 0; l < NL; ++l) {
        gemm_bf16_kernel<false, true, false><<<dim3(MROWS / BM, (3 * DIM) / BN), 256, 0, stream>>>(
            hb, Wqkvb + (size_t)l * DIM * 3 * DIM, bqkv + (size_t)l * 3 * DIM,
            qkvf, nullptr, MROWS, 3 * DIM, DIM);

        rope_split_kernel<<<grid1((size_t)Bn * SEQ * (DIM / 2), 256), 256, 0, stream>>>(
            qkvf, qhb, khb, vhb);

        attention_kernel<<<dim3(Bn * NHD * (SPAD / 128)), 256, 0, stream>>>(
            qhb, khb, vhb, aob);

        gemm_bf16_kernel<false, true, false><<<dim3(MROWS / BM, DIM / BN), 256, 0, stream>>>(
            aob, Wob + (size_t)l * DIM * DIM, bo + (size_t)l * DIM,
            tmpf, nullptr, MROWS, DIM, DIM);

        add_ln_kernel<<<dim3(MROWS), 256, 0, stream>>>(
            hf, tmpf, ln1g + (size_t)l * DIM, ln1b + (size_t)l * DIM, hf, hb);

        gemm_bf16_kernel<true, false, true><<<dim3(MROWS / BM, MLPD / BN), 256, 0, stream>>>(
            hb, Wm1b + (size_t)l * DIM * MLPD, bm1 + (size_t)l * MLPD,
            nullptr, m1b, MROWS, MLPD, DIM);

        gemm_bf16_kernel<false, true, false><<<dim3(MROWS / BM, DIM / BN), 256, 0, stream>>>(
            m1b, Wm2b + (size_t)l * MLPD * DIM, bm2 + (size_t)l * DIM,
            tmpf, nullptr, MROWS, DIM, MLPD);

        add_ln_kernel<<<dim3(MROWS), 256, 0, stream>>>(
            hf, tmpf, ln2g + (size_t)l * DIM, ln2b + (size_t)l * DIM, hf, hb);
    }

    // ---- heads ----
    head_kernel<<<grid1((size_t)Bn * NO * DIM, 256), 256, 0, stream>>>(hf, Wh, bhp, hid);
    varhead_kernel<<<dim3(1), 256, 0, stream>>>(
        hid, mu_wmu, mu_wrho, mu_bmu, mu_brho, lv_wmu, lv_wrho, lv_bmu, lv_brho,
        eps_mu_w, eps_mu_b, eps_lv_w, eps_lv_b, out);
}
